// Decoder_1391569404567
// MI455X (gfx1250) — matrix operand order repaired and checked by reading, not compiled, not run on hardware
//
#include <hip/hip_runtime.h>
#include <hip/hip_bf16.h>

typedef __attribute__((ext_vector_type(16))) _Float16 v16h;
typedef __attribute__((ext_vector_type(8)))  float    v8f;

#define NT     1024
#define CHUNK  128
#define HALO   10
#define EXT    (CHUNK + 2*HALO)   // 148

// ---------------------------------------------------------------------------
// Kernel 1: conv chain on the 2 distinct rows (input repeats 1024x due to the
// tile+reshape index algebra). grid = 2 z_hi * 8 chunks, 256 threads.
// Shrinking-halo trick: edge pollution grows 2/layer * 5 layers = 10 = HALO,
// so the center CHUNK outputs are exact.
// ---------------------------------------------------------------------------
__device__ __forceinline__ void conv_layer(const float (*in)[EXT], float (*out)[EXT],
                                           const float* __restrict__ w,
                                           const float* __restrict__ bias,
                                           int Cin, int Cout, bool relu, int tid)
{
    for (int idx = tid; idx < Cout * EXT; idx += 256) {
        int co = idx / EXT;
        int e  = idx - co * EXT;
        float acc = bias[co];
        for (int ci = 0; ci < Cin; ++ci) {
            const float* wr = w + (co * Cin + ci) * 5;
#pragma unroll
            for (int k = 0; k < 5; ++k) {
                int pos = e + k - 2;
                float xv = (pos >= 0 && pos < EXT) ? in[ci][pos] : 0.0f;
                acc = fmaf(wr[k], xv, acc);
            }
        }
        out[co][e] = relu ? fmaxf(acc, 0.0f) : acc;
    }
}

__global__ void __launch_bounds__(256)
conv_chain_kernel(const float* __restrict__ zl,
                  const float* __restrict__ w1, const float* __restrict__ b1,
                  const float* __restrict__ w2, const float* __restrict__ b2,
                  const float* __restrict__ w3, const float* __restrict__ b3,
                  const float* __restrict__ w4, const float* __restrict__ b4,
                  const float* __restrict__ w5, const float* __restrict__ b5,
                  float* __restrict__ Hflat)          // [2][2][1024] = 4096 f32
{
    __shared__ float bufA[16][EXT];
    __shared__ float bufB[16][EXT];
    __shared__ float wts[1718];

    const int tid   = threadIdx.x;
    const int zhi   = blockIdx.x >> 3;     // 0..1
    const int chunk = blockIdx.x & 7;      // 0..7
    const int start = chunk * CHUNK;

    // stage all weights in LDS:
    // w1:0(160) b1:160(4) w2:164(160) b2:324(8) w3:332(640) b3:972(16)
    // w4:988(640) b4:1628(8) w5:1636(80) b5:1716(2)
    for (int i = tid; i < 160; i += 256) wts[0    + i] = w1[i];
    for (int i = tid; i < 4;   i += 256) wts[160  + i] = b1[i];
    for (int i = tid; i < 160; i += 256) wts[164  + i] = w2[i];
    for (int i = tid; i < 8;   i += 256) wts[324  + i] = b2[i];
    for (int i = tid; i < 640; i += 256) wts[332  + i] = w3[i];
    for (int i = tid; i < 16;  i += 256) wts[972  + i] = b3[i];
    for (int i = tid; i < 640; i += 256) wts[988  + i] = w4[i];
    for (int i = tid; i < 8;   i += 256) wts[1628 + i] = b4[i];
    for (int i = tid; i < 80;  i += 256) wts[1636 + i] = w5[i];
    for (int i = tid; i < 2;   i += 256) wts[1716 + i] = b5[i];

    // load input slab: channel c of distinct row zhi = zl[(zhi*8+c)*1024 + tt]
    for (int idx = tid; idx < 8 * EXT; idx += 256) {
        int c = idx / EXT;
        int e = idx - c * EXT;
        int pos = start - HALO + e;
        bufA[c][e] = (pos >= 0 && pos < NT) ? zl[(zhi * 8 + c) * NT + pos] : 0.0f;
    }
    __syncthreads();

    conv_layer(bufA, bufB, wts + 0,    wts + 160,  8,  4,  true,  tid); __syncthreads();
    conv_layer(bufB, bufA, wts + 164,  wts + 324,  4,  8,  true,  tid); __syncthreads();
    conv_layer(bufA, bufB, wts + 332,  wts + 972,  8,  16, true,  tid); __syncthreads();
    conv_layer(bufB, bufA, wts + 988,  wts + 1628, 16, 8,  true,  tid); __syncthreads();
    conv_layer(bufA, bufB, wts + 1636, wts + 1716, 8,  2,  false, tid); __syncthreads();

    // write valid center: Hflat[zhi][c2][tt]
    for (int idx = tid; idx < 2 * CHUNK; idx += 256) {
        int c2 = idx / CHUNK;
        int e  = idx - c2 * CHUNK;
        Hflat[zhi * 2048 + c2 * NT + (start + e)] = bufB[c2][e + HALO];
    }
}

// ---------------------------------------------------------------------------
// Kernel 2: mean-over-nz collapse + softplus.
// f_mu[t2]   = 0.5*(H0.flat[2t2]   + H1.flat[2t2])
// f_sig[t2]  = 0.5*(H0.flat[2t2+1] + H1.flat[2t2+1]) -> softplus
// ---------------------------------------------------------------------------
__global__ void __launch_bounds__(256)
combine_kernel(const float* __restrict__ Hflat,
               float* __restrict__ Gmu, float* __restrict__ Gsp)
{
    int t2 = blockIdx.x * 256 + threadIdx.x;
    if (t2 < NT) {
        float m = 0.5f * (Hflat[2 * t2]     + Hflat[2048 + 2 * t2]);
        float s = 0.5f * (Hflat[2 * t2 + 1] + Hflat[2048 + 2 * t2 + 1]);
        Gmu[t2] = m;
        Gsp[t2] = (s > 20.0f) ? s : log1pf(__expf(s));   // softplus
    }
}

// ---------------------------------------------------------------------------
// Kernel 3: the real work. D[32768 x 16] = A[32768 x 1024] x B[1024 x 16]
// A generated on the fly: psi(x_i, T) = os * exp(-0.5*((x_i - t_T)/ls)^2)
// B cols: 0 = Gmu, 1 = softplus(Gsig), rest zero (zero-filled LDS source so
// the inner loop is completely branch-free / EXEC-stable for WMMA).
// 1 wave = one 16-row M tile; 32 v_wmma_f32_16x16x32_f16 per tile, unroll x4.
// ---------------------------------------------------------------------------
__global__ void __launch_bounds__(256)
rbf_wmma_kernel(const float* __restrict__ xt, const float* __restrict__ tgrid,
                const float* __restrict__ Gmu, const float* __restrict__ Gsp,
                const float* __restrict__ ls_p, const float* __restrict__ os_p,
                float* __restrict__ out)   // [2][32768] : mu then sigma
{
    __shared__ float sT[NT];
    __shared__ float sMu[NT];
    __shared__ float sSp[NT];
    __shared__ float sZero[NT];

    const int tid = threadIdx.x;
    __builtin_prefetch(Gmu, 0, 0);
    __builtin_prefetch(Gsp, 0, 0);
    for (int i = tid; i < NT; i += 256) {
        sT[i]    = tgrid[i];
        sMu[i]   = Gmu[i];
        sSp[i]   = Gsp[i];
        sZero[i] = 0.0f;
    }
    __syncthreads();

    const float inv_ls = 1.0f / ls_p[0];
    const float os     = os_p[0];

    const int wave = tid >> 5;                 // wave32 on gfx1250
    const int lane = tid & 31;
    const int tile = blockIdx.x * 8 + wave;    // 0..2047 (16 rows each)
    const int g    = lane >> 4;                // lane half-group
    const int M    = lane & 15;                // A row / B column index
    const float x  = xt[tile * 16 + M];

    // branch-free B source: column 0 -> Gmu, column 1 -> softplus(Gsig), else 0
    const float* bsrc = (M == 0) ? sMu : ((M == 1) ? sSp : sZero);
    const float* bsrcg = bsrc + g * 8;         // B: group g holds K = g*8 + h (+8 for h>=8)
    const float* tA    = sT;
    const int    gA8   = g * 8;

    v8f acc = {};
#pragma unroll 4
    for (int kk = 0; kk < NT; kk += 32) {
        v16h a = {};
        v16h b = {};
#pragma unroll
        for (int h = 0; h < 16; ++h) {
            // A fragment: 16-bit A 16x32 layout (ISA 7.12.2):
            // vgpr v=h>>1, half p=h&1 ; K = (v<4?0:16) + g*8 + (v&3)*2 + p
            int v  = h >> 1, p = h & 1;
            int kA = kk + ((v < 4) ? 0 : 16) + gA8 + (v & 3) * 2 + p;
            float d = (x - tA[kA]) * inv_ls;
            a[h] = (_Float16)(os * __expf(-0.5f * d * d));
            // B fragment: unconditional LDS read from per-lane source
            b[h] = (_Float16)bsrcg[kk + h + ((h < 8) ? 0 : 8)];
        }
        acc = __builtin_amdgcn_wmma_f32_16x16x32_f16(
            /*neg_a=*/false, a, /*neg_b=*/false, b,
            /*c_mod=*/(short)0, acc, /*reuse_a=*/false, /*reuse_b=*/false);
    }

    // D layout: lane (N + 16*g) holds column N for rows M = 8*g + v (v = vgpr 0..7).
    // Column 0 -> mu, column 1 -> sigma.
    if (M < 2) {
        float* dst = out + M * 32768;
        int rowbase = tile * 16 + 8 * g;
#pragma unroll
        for (int v = 0; v < 8; ++v) dst[rowbase + v] = acc[v];
    }
}

// ---------------------------------------------------------------------------
extern "C" void kernel_launch(void* const* d_in, const int* in_sizes, int n_in,
                              void* d_out, int out_size, void* d_ws, size_t ws_size,
                              hipStream_t stream) {
    const float* zl = (const float*)d_in[0];
    const float* xt = (const float*)d_in[1];
    const float* t  = (const float*)d_in[2];
    const float* w1 = (const float*)d_in[3];
    const float* b1 = (const float*)d_in[4];
    const float* w2 = (const float*)d_in[5];
    const float* b2 = (const float*)d_in[6];
    const float* w3 = (const float*)d_in[7];
    const float* b3 = (const float*)d_in[8];
    const float* w4 = (const float*)d_in[9];
    const float* b4 = (const float*)d_in[10];
    const float* w5 = (const float*)d_in[11];
    const float* b5 = (const float*)d_in[12];
    const float* ls = (const float*)d_in[13];
    const float* os = (const float*)d_in[14];

    float* ws    = (float*)d_ws;
    float* Hflat = ws;            // 4096 floats
    float* Gmu   = ws + 4096;     // 1024 floats
    float* Gsp   = ws + 5120;     // 1024 floats

    conv_chain_kernel<<<16, 256, 0, stream>>>(zl, w1, b1, w2, b2, w3, b3,
                                              w4, b4, w5, b5, Hflat);
    combine_kernel<<<4, 256, 0, stream>>>(Hflat, Gmu, Gsp);
    rbf_wmma_kernel<<<256, 256, 0, stream>>>(xt, t, Gmu, Gsp, ls, os,
                                             (float*)d_out);
}
